// HGNN_45191645888987
// MI455X (gfx1250) — matrix operand order repaired and checked by reading, not compiled, not run on hardware
//
#include <hip/hip_runtime.h>

// ---- problem constants (from reference) ----
#define F_DIM 300     // feature width
#define DIN   303     // 300 features + 3 coord deltas
#define KP    320     // K padded to multiple of 32 (covers both 303 and 300)
#define NP    320     // N padded to 20 tiles -> 5 tiles per wave-column, branch-free
#define NTILE (NP/16) // 20 column tiles
#define TPW   5       // tiles per wave-column (NTILE / 4), exact
#define AROWS 32      // M tile per block (2 row-groups of 16)
#define ASTR  336     // LDS row stride (bf16 elems); 672B rows, 16B aligned
#define SUM_SCALE 16777216.0   // 2^24 fixed point for sum
#define SQ_SCALE  1048576.0    // 2^20 fixed point for sum of squares

typedef __attribute__((ext_vector_type(16))) __bf16 v16bf;
typedef __attribute__((ext_vector_type(8)))  float  v8f;

union FragBF { uint4 q[2]; v16bf v; };

__device__ __forceinline__ unsigned short f2bf(float f) {
  unsigned u = __float_as_uint(f);
  unsigned r = u + 0x7FFFu + ((u >> 16) & 1u);  // round-to-nearest-even
  return (unsigned short)(r >> 16);
}

// ---------------- K0a: transpose + convert weight to bf16 [NP][KP] ----------------
__global__ void prep_w_kernel(const float* __restrict__ W, unsigned short* __restrict__ Wt,
                              int K, int N) {
  int i = blockIdx.x * blockDim.x + threadIdx.x;
  if (i >= NP * KP) return;
  int n = i / KP, k = i - n * KP;
  float v = (n < N && k < K) ? W[(size_t)k * N + n] : 0.0f;
  Wt[i] = f2bf(v);
}

// ---------------- K0b: init node min/max + zero fixed-point stats ----------------
__global__ void init_kernel(unsigned* __restrict__ nodeMax, unsigned* __restrict__ nodeMin,
                            unsigned long long* __restrict__ stats, int total) {
  int i = blockIdx.x * blockDim.x + threadIdx.x;
  if (i < total) { nodeMax[i] = 0u; nodeMin[i] = 0x7F800000u; }  // +0 / +inf bits
  if (i < 4 * NP) stats[i] = 0ull;
}

// ---------------- K1: fused gather + GEMM1 + bias + ReLU + scatter min/max + stats ----------------
__global__ __launch_bounds__(256)
void edge_gemm_kernel(const float* __restrict__ last_coors,
                      const float* __restrict__ last_features,
                      const float* __restrict__ current_coors,
                      const int* __restrict__ cur_idx,
                      const int* __restrict__ last_idx,
                      const unsigned short* __restrict__ W1t,   // [NP][KP] bf16
                      const float* __restrict__ b1,
                      unsigned* __restrict__ nodeMax,           // [Ncur][NP] f32 bits
                      unsigned* __restrict__ nodeMin,
                      unsigned long long* __restrict__ sum1,    // [NP] fixed point
                      unsigned long long* __restrict__ sq1,
                      int E) {
  __shared__ __align__(16) unsigned short sA[AROWS * ASTR];
  __shared__ int sLast[AROWS];
  __shared__ int sCur[AROWS];

  const int tid = threadIdx.x;
  const int e0  = blockIdx.x * AROWS;

  if (tid < AROWS) {
    int e = e0 + tid;
    sLast[tid] = (e < E) ? last_idx[e] : 0;
    sCur[tid]  = (e < E) ? cur_idx[e]  : 0;
  }
  __syncthreads();

  // gather A tile: 32 rows x KP bf16 (features ++ coord-delta, zero padded)
  for (int i = tid; i < AROWS * KP; i += 256) {
    int row = i / KP, k = i - row * KP;
    int e = e0 + row;
    float v = 0.0f;
    if (e < E) {
      if (k < F_DIM) {
        v = last_features[(size_t)sLast[row] * F_DIM + k];
      } else if (k < DIN) {
        int d = k - F_DIM;
        v = last_coors[sLast[row] * 3 + d] - current_coors[sCur[row] * 3 + d];
      }
    }
    sA[row * ASTR + k] = f2bf(v);
  }
  __syncthreads();

  const int wave = tid >> 5;   // 8 waves
  const int lane = tid & 31;
  const int rg   = wave & 1;   // row-group (M 0..15 / 16..31)
  const int wc   = wave >> 1;  // column owner 0..3
  const int half = lane >> 4;
  const int l16  = lane & 15;

  v8f acc[TPW];
  #pragma unroll
  for (int it = 0; it < TPW; ++it) { v8f z = {0,0,0,0,0,0,0,0}; acc[it] = z; }

  // loop-invariant per-tile B row bases: column n = (wc + it*4)*16 + l16
  const unsigned short* bbase[TPW];
  #pragma unroll
  for (int it = 0; it < TPW; ++it) {
    bbase[it] = W1t + (size_t)((wc + it * 4) * 16 + l16) * KP + half * 8;
  }
  const unsigned short* arow = &sA[(rg * 16 + l16) * ASTR + half * 8];

  for (int kc = 0; kc < KP / 32; ++kc) {
    // A fragment (ISA 16-bit A layout: lanes0-15 K0-7/16-23, lanes16-31 K8-15/24-31)
    FragBF a;
    const unsigned short* ap = arow + kc * 32;
    a.q[0] = *(const uint4*)ap;
    a.q[1] = *(const uint4*)(ap + 16);
    #pragma unroll
    for (int it = 0; it < TPW; ++it) {
      const unsigned short* bp = bbase[it] + kc * 32;
      FragBF b;
      b.q[0] = *(const uint4*)bp;
      b.q[1] = *(const uint4*)(bp + 16);
      __builtin_prefetch(bp + 32, 0, 1);  // next K chunk -> global_prefetch_b8
      acc[it] = __builtin_amdgcn_wmma_f32_16x16x32_bf16(
          false, a.v, false, b.v, (short)0, acc[it], false, false);
    }
  }

  // epilogue: bias + ReLU, scatter max/min per node, fixed-point BN stats
  #pragma unroll
  for (int it = 0; it < TPW; ++it) {
    int n = (wc + it * 4) * 16 + l16;
    if (n >= F_DIM) continue;          // padded columns: acc==0, skip
    float bias = b1[n];
    float ps = 0.0f, pq = 0.0f;
    bool any = false;
    #pragma unroll
    for (int r = 0; r < 8; ++r) {
      int lr = rg * 16 + half * 8 + r;   // C/D layout: M = half*8 + r
      int e = e0 + lr;
      if (e < E) {
        float y = fmaxf(acc[it][r] + bias, 0.0f);
        unsigned yb = __float_as_uint(y);
        unsigned off = (unsigned)sCur[lr] * NP + n;
        atomicMax(nodeMax + off, yb);
        atomicMin(nodeMin + off, yb);
        ps += y; pq += y * y;
        any = true;
      }
    }
    if (any) {
      atomicAdd(sum1 + n, (unsigned long long)((double)ps * SUM_SCALE));
      atomicAdd(sq1  + n, (unsigned long long)((double)pq * SQ_SCALE));
    }
  }
}

// ---------------- K2/K5: BN params from fixed-point stats ----------------
__global__ void bnparam_kernel(const unsigned long long* __restrict__ sum,
                               const unsigned long long* __restrict__ sq,
                               const float* __restrict__ g, const float* __restrict__ be,
                               float* __restrict__ scale, float* __restrict__ shift,
                               float count) {
  int n = blockIdx.x * blockDim.x + threadIdx.x;
  if (n >= F_DIM) return;
  float mean = (float)((double)sum[n] * (1.0 / SUM_SCALE)) / count;
  float ms   = (float)((double)sq[n]  * (1.0 / SQ_SCALE))  / count;
  float var  = ms - mean * mean;
  float s = g[n] * rsqrtf(var + 1e-5f);
  scale[n] = s;
  shift[n] = be[n] - mean * s;
}

// ---------------- K3: resolve segment-max of BN(y) via (max,min), emit bf16 A2 ----------------
__global__ void agg_kernel(const unsigned* __restrict__ nodeMax,
                           const unsigned* __restrict__ nodeMin,
                           const float* __restrict__ scale1, const float* __restrict__ shift1,
                           unsigned short* __restrict__ aggB,  // [Ncur][KP] bf16
                           int Ncur) {
  int i = blockIdx.x * blockDim.x + threadIdx.x;
  if (i >= Ncur * KP) return;
  int node = i / KP, k = i - node * KP;
  float v = 0.0f;
  if (k < F_DIM) {
    unsigned mnb = nodeMin[(size_t)node * NP + k];
    if (mnb != 0x7F800000u) {              // non-empty segment
      float mx = __uint_as_float(nodeMax[(size_t)node * NP + k]);
      float mn = __uint_as_float(mnb);
      float s = scale1[k], sh = shift1[k];
      v = (s >= 0.0f) ? fmaf(s, mx, sh) : fmaf(s, mn, sh);
    }                                      // empty -> 0 (matches isfinite replace)
  }
  aggB[i] = f2bf(v);
}

// ---------------- K4: node GEMM2 + bias + ReLU + stats ----------------
__global__ __launch_bounds__(256)
void node_gemm_kernel(const unsigned short* __restrict__ aggB,  // [Ncur][KP] bf16
                      const unsigned short* __restrict__ W2t,   // [NP][KP] bf16
                      const float* __restrict__ b2,
                      float* __restrict__ zbuf,                 // [Ncur][NP] f32
                      unsigned long long* __restrict__ sum2,
                      unsigned long long* __restrict__ sq2,
                      int Ncur) {
  const int tid = threadIdx.x;
  const int e0  = blockIdx.x * AROWS;
  const int wave = tid >> 5, lane = tid & 31;
  const int rg = wave & 1, wc = wave >> 1;
  const int half = lane >> 4, l16 = lane & 15;

  v8f acc[TPW];
  #pragma unroll
  for (int it = 0; it < TPW; ++it) { v8f z = {0,0,0,0,0,0,0,0}; acc[it] = z; }

  int ar = e0 + rg * 16 + l16;
  if (ar >= Ncur) ar = 0;  // clamp (masked in epilogue)
  const unsigned short* arow = aggB + (size_t)ar * KP + half * 8;

  const unsigned short* bbase[TPW];
  #pragma unroll
  for (int it = 0; it < TPW; ++it) {
    bbase[it] = W2t + (size_t)((wc + it * 4) * 16 + l16) * KP + half * 8;
  }

  for (int kc = 0; kc < KP / 32; ++kc) {
    FragBF a;
    const unsigned short* ap = arow + kc * 32;
    a.q[0] = *(const uint4*)ap;
    a.q[1] = *(const uint4*)(ap + 16);
    #pragma unroll
    for (int it = 0; it < TPW; ++it) {
      const unsigned short* bp = bbase[it] + kc * 32;
      FragBF b;
      b.q[0] = *(const uint4*)bp;
      b.q[1] = *(const uint4*)(bp + 16);
      acc[it] = __builtin_amdgcn_wmma_f32_16x16x32_bf16(
          false, a.v, false, b.v, (short)0, acc[it], false, false);
    }
  }

  #pragma unroll
  for (int it = 0; it < TPW; ++it) {
    int n = (wc + it * 4) * 16 + l16;
    if (n >= F_DIM) continue;
    float bias = b2[n];
    float ps = 0.0f, pq = 0.0f;
    bool any = false;
    #pragma unroll
    for (int r = 0; r < 8; ++r) {
      int node = e0 + rg * 16 + half * 8 + r;
      if (node < Ncur) {
        float z = fmaxf(acc[it][r] + bias, 0.0f);
        zbuf[(size_t)node * NP + n] = z;
        ps += z; pq += z * z;
        any = true;
      }
    }
    if (any) {
      atomicAdd(sum2 + n, (unsigned long long)((double)ps * SUM_SCALE));
      atomicAdd(sq2  + n, (unsigned long long)((double)pq * SQ_SCALE));
    }
  }
}

// ---------------- K6: apply BN2 -> output [Ncur][300] ----------------
__global__ void bnapply_kernel(const float* __restrict__ zbuf,
                               const float* __restrict__ scale2, const float* __restrict__ shift2,
                               float* __restrict__ out, int Ncur) {
  int i = blockIdx.x * blockDim.x + threadIdx.x;
  if (i >= Ncur * F_DIM) return;
  int node = i / F_DIM, n = i - node * F_DIM;
  out[i] = fmaf(zbuf[(size_t)node * NP + n], scale2[n], shift2[n]);
}

extern "C" void kernel_launch(void* const* d_in, const int* in_sizes, int n_in,
                              void* d_out, int out_size, void* d_ws, size_t ws_size,
                              hipStream_t stream) {
  const float* last_coors    = (const float*)d_in[0];
  const float* last_features = (const float*)d_in[1];
  const float* current_coors = (const float*)d_in[2];
  const int*   cur_idx       = (const int*)d_in[3];
  const int*   last_idx      = (const int*)d_in[4];
  const float* W1  = (const float*)d_in[5];
  const float* b1  = (const float*)d_in[6];
  const float* g1  = (const float*)d_in[7];
  const float* be1 = (const float*)d_in[8];
  const float* W2  = (const float*)d_in[9];
  const float* b2  = (const float*)d_in[10];
  const float* g2  = (const float*)d_in[11];
  const float* be2 = (const float*)d_in[12];

  const int E    = in_sizes[3];
  const int Ncur = in_sizes[2] / 3;

  // workspace carve-up (all chunks 16B+ aligned)
  char* ws = (char*)d_ws;
  unsigned short* W1t = (unsigned short*)ws;  ws += (size_t)NP * KP * 2;
  unsigned short* W2t = (unsigned short*)ws;  ws += (size_t)NP * KP * 2;
  unsigned long long* stats = (unsigned long long*)ws;  ws += (size_t)4 * NP * 8;
  float* bnp = (float*)ws;                    ws += (size_t)4 * NP * 4;
  unsigned* nodeMax = (unsigned*)ws;          ws += (size_t)Ncur * NP * 4;
  unsigned* nodeMin = (unsigned*)ws;          ws += (size_t)Ncur * NP * 4;
  unsigned short* aggB = (unsigned short*)ws; ws += (size_t)Ncur * KP * 2;
  float* zbuf = (float*)nodeMax;  // safe reuse: nodeMax consumed before K4 runs

  unsigned long long* sum1 = stats;
  unsigned long long* sq1  = stats + NP;
  unsigned long long* sum2 = stats + 2 * NP;
  unsigned long long* sq2  = stats + 3 * NP;
  float* scale1 = bnp;          float* shift1 = bnp + NP;
  float* scale2 = bnp + 2*NP;   float* shift2 = bnp + 3*NP;

  // K0: weight prep + init
  prep_w_kernel<<<(NP * KP + 255) / 256, 256, 0, stream>>>(W1, W1t, DIN, F_DIM);
  prep_w_kernel<<<(NP * KP + 255) / 256, 256, 0, stream>>>(W2, W2t, F_DIM, F_DIM);
  init_kernel<<<((size_t)Ncur * NP + 255) / 256, 256, 0, stream>>>(nodeMax, nodeMin, stats, Ncur * NP);

  // K1: edge pass (fused gather + GEMM1 + ReLU + scatter min/max + stats)
  edge_gemm_kernel<<<(E + AROWS - 1) / AROWS, 256, 0, stream>>>(
      last_coors, last_features, current_coors, cur_idx, last_idx,
      W1t, b1, nodeMax, nodeMin, sum1, sq1, E);

  // K2: BN1 params
  bnparam_kernel<<<(F_DIM + 255) / 256, 256, 0, stream>>>(sum1, sq1, g1, be1, scale1, shift1, (float)E);

  // K3: resolve aggregation, emit bf16 A for GEMM2
  agg_kernel<<<((size_t)Ncur * KP + 255) / 256, 256, 0, stream>>>(nodeMax, nodeMin, scale1, shift1, aggB, Ncur);

  // K4: node GEMM2
  node_gemm_kernel<<<(Ncur + AROWS - 1) / AROWS, 256, 0, stream>>>(aggB, W2t, b2, zbuf, sum2, sq2, Ncur);

  // K5: BN2 params
  bnparam_kernel<<<(F_DIM + 255) / 256, 256, 0, stream>>>(sum2, sq2, g2, be2, scale2, shift2, (float)Ncur);

  // K6: apply BN2 -> out
  bnapply_kernel<<<((size_t)Ncur * F_DIM + 255) / 256, 256, 0, stream>>>(zbuf, scale2, shift2, (float*)d_out, Ncur);

  (void)n_in; (void)out_size; (void)ws_size;
}
